// PCEncoder_34909494182010
// MI455X (gfx1250) — matrix operand order repaired and checked by reading, not compiled, not run on hardware
//
#include <hip/hip_runtime.h>
#include <hip/hip_bf16.h>

typedef __attribute__((ext_vector_type(16))) _Float16 v16h;
typedef __attribute__((ext_vector_type(8)))  float    v8f;

// ---------------------------------------------------------------------------
// Farthest point sampling: one workgroup per batch, dist array in LDS (32KB),
// parallel argmax with first-index tie-break (matches jnp.argmax).
// Writes only new_xyz (the indices themselves are never needed downstream).
// ---------------------------------------------------------------------------
__global__ __launch_bounds__(256) void fps_kernel(const float* __restrict__ xyz,
                                                  int N, int npoint,
                                                  float* __restrict__ newXyz) {
  __shared__ float dist[8192];
  __shared__ float redV[256];
  __shared__ int   redI[256];
  __shared__ float selPt[3];
  const int b = blockIdx.x;
  const int t = threadIdx.x;
  const float* X = xyz + (size_t)b * N * 3;
  float* NX = newXyz + (size_t)b * npoint * 3;
  const float x0 = X[0], y0 = X[1], z0 = X[2];
  for (int i = t; i < N; i += 256) {
    float dx = X[i*3+0]-x0, dy = X[i*3+1]-y0, dz = X[i*3+2]-z0;
    dist[i] = dx*dx + dy*dy + dz*dz;
  }
  if (t == 0) { NX[0] = x0; NX[1] = y0; NX[2] = z0; }
  __syncthreads();
  for (int it = 1; it < npoint; ++it) {
    float bv = -1.0f; int bi = 0;
    for (int i = t; i < N; i += 256) {
      float v = dist[i];
      if (v > bv) { bv = v; bi = i; }       // strict > keeps lowest index in-thread
    }
    redV[t] = bv; redI[t] = bi;
    __syncthreads();
    for (int s = 128; s > 0; s >>= 1) {
      if (t < s) {
        float v2 = redV[t+s]; int i2 = redI[t+s];
        if (v2 > redV[t] || (v2 == redV[t] && i2 < redI[t])) { redV[t] = v2; redI[t] = i2; }
      }
      __syncthreads();
    }
    if (t == 0) {
      int ni = redI[0];
      float px = X[ni*3+0], py = X[ni*3+1], pz = X[ni*3+2];
      selPt[0] = px; selPt[1] = py; selPt[2] = pz;
      NX[it*3+0] = px; NX[it*3+1] = py; NX[it*3+2] = pz;
    }
    __syncthreads();
    float cx = selPt[0], cy = selPt[1], cz = selPt[2];
    for (int i = t; i < N; i += 256) {
      float dx = X[i*3+0]-cx, dy = X[i*3+1]-cy, dz = X[i*3+2]-cz;
      dist[i] = fminf(dist[i], dx*dx + dy*dy + dz*dz);
    }
    __syncthreads();
  }
}

// ---------------------------------------------------------------------------
// Ball query: first nsample(=64) in-ball indices in ascending order, padded
// with the first hit (or 0). One thread per query; xyz staged via LDS chunks.
// ---------------------------------------------------------------------------
#define BQ_CHUNK 2048
__global__ __launch_bounds__(64) void ball_query_kernel(const float* __restrict__ xyz,
                                                        const float* __restrict__ newXyz,
                                                        int N, int npoint, float r2,
                                                        int* __restrict__ outIdx) {
  __shared__ float sp[BQ_CHUNK * 3];
  const int b = blockIdx.x;
  const int q = blockIdx.y * 64 + threadIdx.x;
  float qx = 0.f, qy = 0.f, qz = 0.f;
  if (q < npoint) {
    const float* nq = newXyz + ((size_t)b * npoint + q) * 3;
    qx = nq[0]; qy = nq[1]; qz = nq[2];
  }
  int cnt = 0; int first = 0;
  int* dst = outIdx + ((size_t)b * npoint + q) * 64;
  for (int c0 = 0; c0 < N; c0 += BQ_CHUNK) {
    int cn = min(BQ_CHUNK, N - c0);
    __syncthreads();
    for (int i = threadIdx.x; i < cn * 3; i += 64)
      sp[i] = xyz[((size_t)b * N + c0) * 3 + i];
    __syncthreads();
    if (q < npoint) {
      for (int i = 0; i < cn; ++i) {
        float dx = sp[i*3+0]-qx, dy = sp[i*3+1]-qy, dz = sp[i*3+2]-qz;
        if (dx*dx + dy*dy + dz*dz < r2) {
          if (cnt == 0) first = c0 + i;
          if (cnt < 64) dst[cnt] = c0 + i;
          ++cnt;
        }
      }
    }
  }
  if (q < npoint) {
    int c = cnt < 64 ? cnt : 64;
    for (int k = c; k < 64; ++k) dst[k] = first;   // first==0 when empty
  }
}

// ---------------------------------------------------------------------------
// Grouping / concat kernels producing f16 GEMM inputs (K zero-padded).
// ---------------------------------------------------------------------------
__global__ void gather1_kernel(const float* __restrict__ xyz,
                               const float* __restrict__ newXyz,
                               const int* __restrict__ ballIdx,
                               _Float16* __restrict__ X) {
  // rows = 32*512*64, Kp = 32; cols 0..2 rel xyz, 3..5 abs xyz (feats==xyz)
  long i = (long)blockIdx.x * 256 + threadIdx.x;
  long total = (long)32 * 512 * 64 * 32;
  if (i >= total) return;
  int col = (int)(i & 31); long m = i >> 5;
  long bj = m >> 6; int b = (int)(bj >> 9);
  _Float16 v = (_Float16)0.f;
  if (col < 6) {
    int idx = ballIdx[m];
    const float* p = xyz + ((long)b * 8192 + idx) * 3;
    if (col < 3) v = (_Float16)(p[col] - newXyz[bj*3 + col]);
    else         v = (_Float16)p[col - 3];
  }
  X[i] = v;
}

__global__ void gather2_kernel(const float* __restrict__ xyz1,
                               const float* __restrict__ newXyz2,
                               const int* __restrict__ ballIdx,
                               const float* __restrict__ feats1,
                               _Float16* __restrict__ X) {
  // rows = 32*256*64, Kp = 160; cols 0..2 rel xyz, 3..130 feats1(128)
  long i = (long)blockIdx.x * 256 + threadIdx.x;
  long total = (long)32 * 256 * 64 * 160;
  if (i >= total) return;
  int col = (int)(i % 160); long m = i / 160;
  long bj = m >> 6; int b = (int)(bj >> 8);
  _Float16 v = (_Float16)0.f;
  int idx = ballIdx[m];
  if (col < 3)       v = (_Float16)(xyz1[((long)b*512 + idx)*3 + col] - newXyz2[bj*3 + col]);
  else if (col < 131) v = (_Float16)feats1[((long)b*512 + idx)*128 + (col - 3)];
  X[i] = v;
}

__global__ void gather3_kernel(const float* __restrict__ xyz2,
                               const float* __restrict__ feats2,
                               _Float16* __restrict__ X) {
  // rows = 32*256, Kp = 288; cols 0..2 xyz2, 3..258 feats2(256)
  long i = (long)blockIdx.x * 256 + threadIdx.x;
  long total = (long)8192 * 288;
  if (i >= total) return;
  int col = (int)(i % 288); long m = i / 288;
  _Float16 v = (_Float16)0.f;
  if (col < 3)        v = (_Float16)xyz2[m*3 + col];
  else if (col < 259) v = (_Float16)feats2[m*256 + (col - 3)];
  X[i] = v;
}

// ---------------------------------------------------------------------------
// Weight prep: f32 W[Cin x Cout] -> f16 Wt[Cout x Kp] (transposed, K-padded).
// ---------------------------------------------------------------------------
__global__ void wconv_kernel(const float* __restrict__ W, int Cin, int Cout, int Kp,
                             _Float16* __restrict__ Wt) {
  long i = (long)blockIdx.x * 256 + threadIdx.x;
  if (i >= (long)Cout * Kp) return;
  int k = (int)(i % Kp); int o = (int)(i / Kp);
  Wt[i] = (k < Cin) ? (_Float16)W[(long)k * Cout + o] : (_Float16)0.f;
}

// ---------------------------------------------------------------------------
// WMMA GEMM: Y[M x C] = A[M x Kp] * Wt^T  (f16 in, f32 accum, f16 out, no bias
// -- bias cancels in batch-norm). 4 waves/block, one 16x16 tile per wave.
// Fragment loads follow the CDNA5 wave32 VGPR layouts (ISA 7.12.2).
// ---------------------------------------------------------------------------
union HalfQ { uint4 q; _Float16 h[8]; };

__global__ __launch_bounds__(128) void gemm_wmma_kernel(const _Float16* __restrict__ A,
                                                        const _Float16* __restrict__ Wt,
                                                        _Float16* __restrict__ Y,
                                                        int Kp, int C) {
  const int lane = threadIdx.x & 31;
  const int wave = threadIdx.x >> 5;
  const long rowBase = ((long)blockIdx.x * 4 + wave) * 16;
  const int  colBase = blockIdx.y * 16;
  const int  h  = lane >> 4;     // half-wave select
  const int  ml = lane & 15;     // A: row-in-tile; B: col-in-tile

  const _Float16* arow = A  + (rowBase + ml) * (long)Kp;
  const _Float16* brow = Wt + (long)(colBase + ml) * Kp;

  v8f c = {};
  for (int k0 = 0; k0 < Kp; k0 += 32) {
    // A fragment (16x32 f16): lanes 0-15 K=[0,8)+[16,24), lanes 16-31 K=[8,16)+[24,32)
    const _Float16* p1 = arow + k0 + h * 8;
    HalfQ ua0, ua1;
    ua0.q = *(const uint4*)(p1);
    ua1.q = *(const uint4*)(p1 + 16);
    // B fragment (32x16 f16): lane owns col ml, K=[8h,8h+8)+[16+8h,24+8h)
    const _Float16* pb = brow + k0 + h * 8;
    HalfQ ub0, ub1;
    ub0.q = *(const uint4*)(pb);
    ub1.q = *(const uint4*)(pb + 16);
    if (k0 + 32 < Kp) __builtin_prefetch((const void*)(p1 + 32), 0, 3); // global_prefetch_b8

    v16h a, b;
#pragma unroll
    for (int i = 0; i < 8; ++i) { a[i] = ua0.h[i]; a[8 + i] = ua1.h[i]; }
#pragma unroll
    for (int i = 0; i < 8; ++i) { b[i] = ub0.h[i]; b[8 + i] = ub1.h[i]; }

    c = __builtin_amdgcn_wmma_f32_16x16x32_f16(false, a, false, b,
                                               (short)0, c, false, false);
  }
  // C/D layout: n = lane%16, m = r + 8*(lane/16)
#pragma unroll
  for (int r = 0; r < 8; ++r)
    Y[(rowBase + r + 8*h) * (long)C + colBase + ml] = (_Float16)c[r];
}

// ---------------------------------------------------------------------------
// Per-channel sum / sum-of-squares over the raw f16 GEMM output (BN stats).
// ---------------------------------------------------------------------------
__global__ __launch_bounds__(256) void colsum_kernel(const _Float16* __restrict__ Y,
                                                     long M, int C, int rowsPerBlock,
                                                     float* __restrict__ sum,
                                                     float* __restrict__ sumsq) {
  long r0 = (long)blockIdx.x * rowsPerBlock;
  long r1 = r0 + rowsPerBlock; if (r1 > M) r1 = M;
  float acc[4]  = {0.f, 0.f, 0.f, 0.f};
  float acc2[4] = {0.f, 0.f, 0.f, 0.f};
  for (long r = r0; r < r1; ++r) {
    const _Float16* row = Y + r * (long)C;
    int k = 0;
    for (int c = threadIdx.x; c < C; c += 256, ++k) {
      float v = (float)row[c];
      acc[k] += v; acc2[k] += v * v;
    }
  }
  int k = 0;
  for (int c = threadIdx.x; c < C; c += 256, ++k) {
    atomicAdd(&sum[c],   acc[k]);
    atomicAdd(&sumsq[c], acc2[k]);
  }
}

__global__ void zero_kernel(float* __restrict__ p, int n) {
  int i = blockIdx.x * 256 + threadIdx.x;
  if (i < n) p[i] = 0.f;
}

__global__ void bn_finalize_kernel(const float* __restrict__ sum,
                                   const float* __restrict__ sumsq,
                                   const float* __restrict__ g,
                                   const float* __restrict__ be,
                                   float invM, int C, float* __restrict__ ss) {
  int c = blockIdx.x * 64 + threadIdx.x;
  if (c >= C) return;
  float mean = sum[c] * invM;
  float var  = sumsq[c] * invM - mean * mean;
  float sc   = g[c] * rsqrtf(var + 1e-5f);
  ss[c]     = sc;
  ss[C + c] = be[c] - mean * sc;
}

__global__ __launch_bounds__(256) void bn_relu_kernel(_Float16* __restrict__ Y,
                                                      const float* __restrict__ ss,
                                                      long total, int C) {
  long i8 = ((long)blockIdx.x * 256 + threadIdx.x) * 8;
  if (i8 >= total) return;
  int c = (int)(i8 % C);
#pragma unroll
  for (int j = 0; j < 8; ++j) {
    float v = (float)Y[i8 + j] * ss[c + j] + ss[C + c + j];
    Y[i8 + j] = (_Float16)fmaxf(v, 0.f);
  }
}

// ---------------------------------------------------------------------------
// FiLM parameter GEMV (betas[32x16] @ W[16xC] + b) and max-pool + FiLM.
// ---------------------------------------------------------------------------
__global__ void film_params_kernel(const float* __restrict__ betas,
                                   const float* __restrict__ Wf, const float* __restrict__ bf,
                                   const float* __restrict__ Wh, const float* __restrict__ bh,
                                   int C, float* __restrict__ gamma, float* __restrict__ beta) {
  int i = blockIdx.x * 256 + threadIdx.x;
  if (i >= 32 * C) return;
  int c = i % C, b = i / C;
  float g = bf[c], hh = bh[c];
  for (int k = 0; k < 16; ++k) {
    float bk = betas[b * 16 + k];
    g  += bk * Wf[k * C + c];
    hh += bk * Wh[k * C + c];
  }
  gamma[i] = g; beta[i] = hh;
}

__global__ __launch_bounds__(256) void maxpool_film_kernel(const _Float16* __restrict__ Y,
                                                           const float* __restrict__ gamma,
                                                           const float* __restrict__ beta,
                                                           int np, int ns, int C,
                                                           float* __restrict__ out) {
  long o = (long)blockIdx.x * 256 + threadIdx.x;
  long total = (long)32 * np * C;
  if (o >= total) return;
  int c = (int)(o % C); long bj = o / C; int b = (int)(bj / np);
  const _Float16* p = Y + bj * (long)ns * C + c;
  float m = -3.4e38f;
  for (int s = 0; s < ns; ++s) m = fmaxf(m, (float)p[(long)s * C]);
  out[o] = m * gamma[b * C + c] + beta[b * C + c];
}

__global__ void fc_kernel(const float* __restrict__ f, const float* __restrict__ W,
                          const float* __restrict__ bias, float* __restrict__ out) {
  int i = blockIdx.x * 256 + threadIdx.x;
  if (i >= 32 * 256) return;
  int o = i % 256, b = i / 256;
  float acc = bias[o];
  const float* fr = f + b * 1024;
  for (int c = 0; c < 1024; ++c) acc += fr[c] * W[c * 256 + o];
  out[i] = acc;
}

// ---------------------------------------------------------------------------
// Orchestration
// ---------------------------------------------------------------------------
extern "C" void kernel_launch(void* const* d_in, const int* in_sizes, int n_in,
                              void* d_out, int out_size, void* d_ws, size_t ws_size,
                              hipStream_t stream) {
  (void)in_sizes; (void)n_in; (void)out_size; (void)ws_size;

  const float* pc    = (const float*)d_in[0];
  const float* betas = (const float*)d_in[1];

  // layer i (0..8): W = d_in[2+4i], (bias unused), g = +2, be = +3
  const float *W_[9], *G_[9], *Be_[9];
  for (int i = 0; i < 9; ++i) {
    W_[i]  = (const float*)d_in[2 + 4*i];
    G_[i]  = (const float*)d_in[2 + 4*i + 2];
    Be_[i] = (const float*)d_in[2 + 4*i + 3];
  }
  const int Cin[9]  = {6, 64, 64,   131, 128, 128,  259, 256, 512};
  const int Cout[9] = {64, 64, 128, 128, 128, 256,  256, 512, 1024};
  const int Kp[9]   = {32, 64, 64,  160, 128, 128,  288, 256, 512};

  // deterministic bump allocator over d_ws
  char* base = (char*)d_ws; size_t off = 0;
  auto wsAlloc = [&](size_t bytes) -> void* {
    void* p = base + off; off += (bytes + 255) & ~(size_t)255; return p;
  };
  float* newXyz1 = (float*)wsAlloc((size_t)32*512*3*4);
  float* newXyz2 = (float*)wsAlloc((size_t)32*256*3*4);
  int*   ballIdx1 = (int*)wsAlloc((size_t)32*512*64*4);
  int*   ballIdx2 = (int*)wsAlloc((size_t)32*256*64*4);
  float* feats1 = (float*)wsAlloc((size_t)32*512*128*4);
  float* feats2 = (float*)wsAlloc((size_t)32*256*256*4);
  float* feats3 = (float*)wsAlloc((size_t)32*1024*4);
  float* stats  = (float*)wsAlloc((size_t)2048*4);     // [sum(1024) | sumsq(1024)]
  float* ss     = (float*)wsAlloc((size_t)2048*4);     // [scale(C) | shift(C)]
  float* gamma1 = (float*)wsAlloc((size_t)32*128*4);
  float* beta1  = (float*)wsAlloc((size_t)32*128*4);
  float* gamma2 = (float*)wsAlloc((size_t)32*256*4);
  float* beta2  = (float*)wsAlloc((size_t)32*256*4);
  float* gamma3 = (float*)wsAlloc((size_t)32*1024*4);
  float* beta3  = (float*)wsAlloc((size_t)32*1024*4);
  _Float16* Wt[9];
  for (int i = 0; i < 9; ++i) Wt[i] = (_Float16*)wsAlloc((size_t)Cout[i]*Kp[i]*2);
  _Float16* bufA = (_Float16*)wsAlloc((size_t)192*1024*1024); // X1/X2/Y(b) ping
  _Float16* bufB = (_Float16*)wsAlloc((size_t)256*1024*1024); // Y(a)/Y(c) pong

  // weight transposition / f16 conversion
  for (int i = 0; i < 9; ++i) {
    long n = (long)Cout[i]*Kp[i];
    wconv_kernel<<<(unsigned)((n + 255)/256), 256, 0, stream>>>(W_[i], Cin[i], Cout[i], Kp[i], Wt[i]);
  }
  // FiLM params
  film_params_kernel<<<(32*128 + 255)/256, 256, 0, stream>>>(
      betas, (const float*)d_in[38], (const float*)d_in[39],
      (const float*)d_in[40], (const float*)d_in[41], 128, gamma1, beta1);
  film_params_kernel<<<(32*256 + 255)/256, 256, 0, stream>>>(
      betas, (const float*)d_in[42], (const float*)d_in[43],
      (const float*)d_in[44], (const float*)d_in[45], 256, gamma2, beta2);
  film_params_kernel<<<(32*1024 + 255)/256, 256, 0, stream>>>(
      betas, (const float*)d_in[46], (const float*)d_in[47],
      (const float*)d_in[48], (const float*)d_in[49], 1024, gamma3, beta3);

  auto runLayer = [&](const _Float16* Xin, _Float16* Yout, long M, int kp, int C,
                      const _Float16* wt, const float* g, const float* be) {
    zero_kernel<<<8, 256, 0, stream>>>(stats, 2048);
    dim3 gg((unsigned)(M / 64), (unsigned)(C / 16));
    gemm_wmma_kernel<<<gg, 128, 0, stream>>>(Xin, wt, Yout, kp, C);
    long nb = (M + 255) / 256;
    colsum_kernel<<<(unsigned)nb, 256, 0, stream>>>(Yout, M, C, 256, stats, stats + 1024);
    bn_finalize_kernel<<<(C + 63)/64, 64, 0, stream>>>(stats, stats + 1024, g, be,
                                                       1.0f / (float)M, C, ss);
    long tot = M * (long)C;
    bn_relu_kernel<<<(unsigned)((tot/8 + 255)/256), 256, 0, stream>>>(Yout, ss, tot, C);
  };

  // -------- SA1: N=8192 -> np=512, ns=64, r=0.2, MLP 6->64->64->128 --------
  fps_kernel<<<32, 256, 0, stream>>>(pc, 8192, 512, newXyz1);
  ball_query_kernel<<<dim3(32, 8), 64, 0, stream>>>(pc, newXyz1, 8192, 512, 0.04f, ballIdx1);
  {
    long tot = (long)32*512*64*32;
    gather1_kernel<<<(unsigned)((tot + 255)/256), 256, 0, stream>>>(pc, newXyz1, ballIdx1, bufA);
  }
  const long M1 = (long)32*512*64;
  runLayer(bufA, bufB, M1, Kp[0], Cout[0], Wt[0], G_[0], Be_[0]);
  runLayer(bufB, bufA, M1, Kp[1], Cout[1], Wt[1], G_[1], Be_[1]);
  runLayer(bufA, bufB, M1, Kp[2], Cout[2], Wt[2], G_[2], Be_[2]);
  maxpool_film_kernel<<<(unsigned)(((long)32*512*128 + 255)/256), 256, 0, stream>>>(
      bufB, gamma1, beta1, 512, 64, 128, feats1);

  // -------- SA2: N=512 -> np=256, ns=64, r=0.4, MLP 131->128->128->256 -----
  fps_kernel<<<32, 256, 0, stream>>>(newXyz1, 512, 256, newXyz2);
  ball_query_kernel<<<dim3(32, 4), 64, 0, stream>>>(newXyz1, newXyz2, 512, 256, 0.16f, ballIdx2);
  {
    long tot = (long)32*256*64*160;
    gather2_kernel<<<(unsigned)((tot + 255)/256), 256, 0, stream>>>(newXyz1, newXyz2, ballIdx2,
                                                                    feats1, bufA);
  }
  const long M2 = (long)32*256*64;
  runLayer(bufA, bufB, M2, Kp[3], Cout[3], Wt[3], G_[3], Be_[3]);
  runLayer(bufB, bufA, M2, Kp[4], Cout[4], Wt[4], G_[4], Be_[4]);
  runLayer(bufA, bufB, M2, Kp[5], Cout[5], Wt[5], G_[5], Be_[5]);
  maxpool_film_kernel<<<(unsigned)(((long)32*256*256 + 255)/256), 256, 0, stream>>>(
      bufB, gamma2, beta2, 256, 64, 256, feats2);

  // -------- SA3 (group-all): 256 pts, MLP 259->256->512->1024 --------------
  {
    long tot = (long)8192*288;
    gather3_kernel<<<(unsigned)((tot + 255)/256), 256, 0, stream>>>(newXyz2, feats2, bufA);
  }
  const long M3 = 8192;
  runLayer(bufA, bufB, M3, Kp[6], Cout[6], Wt[6], G_[6], Be_[6]);
  runLayer(bufB, bufA, M3, Kp[7], Cout[7], Wt[7], G_[7], Be_[7]);
  runLayer(bufA, bufB, M3, Kp[8], Cout[8], Wt[8], G_[8], Be_[8]);
  maxpool_film_kernel<<<(32*1024 + 255)/256, 256, 0, stream>>>(
      bufB, gamma3, beta3, 1, 256, 1024, feats3);

  // -------- FC head: (32,1024) @ (1024,256) + b -> d_out -------------------
  fc_kernel<<<(32*256 + 255)/256, 256, 0, stream>>>(
      feats3, (const float*)d_in[50], (const float*)d_in[51], (float*)d_out);
}
